// GNNGraphColoring_36223754174949
// MI455X (gfx1250) — compile-verified, matrix operand order checked by reading, not run on hardware
//
#include <hip/hip_runtime.h>
#include <hip/hip_bf16.h>

typedef __attribute__((ext_vector_type(16))) _Float16 v16h;
typedef __attribute__((ext_vector_type(8)))  _Float16 v8h;
typedef __attribute__((ext_vector_type(8)))  float    v8f;

#define FIN 128
#define HID 128
#define COUT 16
#define LDK 136        // padded K stride (halves): 272B = 68 dwords -> conflict-free, 16B aligned
#define TILES_PB 10    // row tiles (of 16) per gemm1 block: 160 rows/block

// Build a 16xf16 WMMA fragment from two contiguous 8xf16 runs (each one
// vector load: ds_load_b128 / global_load_b128).
__device__ __forceinline__ v16h frag16(const _Float16* p_lo, const _Float16* p_hi) {
    v8h lo = *(const v8h*)p_lo;
    v8h hi = *(const v8h*)p_hi;
    return __builtin_shufflevector(lo, hi, 0,1,2,3,4,5,6,7,8,9,10,11,12,13,14,15);
}

// ---------------------------------------------------------------------------
// Degree / normalization kernels
// ---------------------------------------------------------------------------
__global__ __launch_bounds__(256) void deg_init_kernel(float* deg, int n) {
    int i = blockIdx.x * blockDim.x + threadIdx.x;
    if (i < n) deg[i] = 1.0f;   // self-loop contributes 1 to every node's degree
}

__global__ __launch_bounds__(256) void deg_count_kernel(const long long* dst, float* deg, long long E) {
    long long e = (long long)blockIdx.x * blockDim.x + threadIdx.x;
    if (e < E) atomicAdd(&deg[(int)dst[e]], 1.0f);
}

__global__ __launch_bounds__(256) void dis_kernel(float* deg_dis, int n) {
    int i = blockIdx.x * blockDim.x + threadIdx.x;
    if (i < n) deg_dis[i] = rsqrtf(deg_dis[i]);   // deg >= 1 always
}

// ---------------------------------------------------------------------------
// GEMM1: h1[N,128] = x[N,128] @ W1[128,128], f16 WMMA / f32 accumulate.
// Block = 256 threads (8 waves). W1 is staged TRANSPOSED in LDS ONCE, then the
// block loops over TILES_PB 16-row tiles of x with a ping-pong x-tile buffer
// (one barrier per tile). Within a tile, wave w owns output-column tile w*16.
// ---------------------------------------------------------------------------
__global__ __launch_bounds__(256) void gemm1_kernel(const float* __restrict__ x,
                                                    const float* __restrict__ W1,
                                                    float* __restrict__ h1, int n) {
    __shared__ _Float16 wst[HID * LDK];     // transposed W1: wst[n*LDK + k], ~34 KB
    __shared__ _Float16 xs[2][16 * LDK];    // ping-pong x tiles, ~8.7 KB
    const int tid = threadIdx.x;

    // Stage W1 transposed, once per block. At each loop step the 128 threads
    // of a half-block read 128 consecutive floats (coalesced); each thread's
    // LDS writes are contiguous halves (vectorizable).
    {
        const int c  = tid & 127;            // output column
        const int k0 = (tid >> 7) * 64;      // half the K range per thread
        for (int k = k0; k < k0 + 64; ++k)
            wst[c * LDK + k] = (_Float16)W1[(size_t)k * HID + c];
    }

    const int wave = tid >> 5;
    const int lane = tid & 31;
    const int n0 = wave * 16;              // this wave's output-column tile
    const int m  = lane & 15;              // A row held by this lane
    const int nc = lane & 15;              // B/C column held by this lane
    const int kb = (lane >> 4) * 8;        // half-wave K sub-offset
    const _Float16* bcol = &wst[(n0 + nc) * LDK];

    // Per-thread x-tile staging slice: 8 contiguous floats -> 8 contiguous halves.
    const int xr = (tid * 8) >> 7;         // row within tile (0..15)
    const int xc = (tid * 8) & 127;        // col start (multiple of 8 -> 16B aligned)

    const int rowBase = blockIdx.x * (16 * TILES_PB);

    for (int t = 0; t < TILES_PB; ++t) {
        const int rowTile = rowBase + t * 16;
        _Float16* xbuf = xs[t & 1];
        if (rowTile < n) {
            const float* xg = &x[(size_t)(rowTile + xr) * FIN + xc];
            _Float16*    xd = &xbuf[xr * LDK + xc];
#pragma unroll
            for (int j = 0; j < 8; ++j) xd[j] = (_Float16)xg[j];
        }
        __syncthreads();   // xbuf staged; previous tile's compute (other buffer) done
        if (rowTile < n) {
            const _Float16* arow = &xbuf[m * LDK];
            v8f acc = {};
#pragma unroll
            for (int k0 = 0; k0 < FIN; k0 += 32) {
                v16h a = frag16(arow + k0 + kb, arow + k0 + 16 + kb);
                v16h b = frag16(bcol + k0 + kb, bcol + k0 + 16 + kb);
                acc = __builtin_amdgcn_wmma_f32_16x16x32_f16(false, a, false, b,
                                                             (short)0, acc, false, false);
            }
#pragma unroll
            for (int r = 0; r < 8; ++r) {
                int mr = r + (lane >> 4) * 8;   // C/D: VGPR r -> M=r (lanes<16) / M=8+r
                h1[(size_t)(rowTile + mr) * HID + n0 + nc] = acc[r];
            }
        }
    }
}

// ---------------------------------------------------------------------------
// Aggregation layer 1: agg = sum_{e: dst=i} h1[src]*dis[src]*dis[dst]
// Init folds in the self-loop term and bias: agg[i] = h1[i]*dis[i]^2 + b1.
// ---------------------------------------------------------------------------
__global__ __launch_bounds__(256) void agg1_init_kernel(const float* __restrict__ h1,
                                                        const float* __restrict__ dis,
                                                        const float* __restrict__ b1,
                                                        float* __restrict__ agg, int n) {
    int gid = blockIdx.x * blockDim.x + threadIdx.x;
    if (gid < n * HID) {
        int i = gid >> 7, f = gid & 127;
        float d = dis[i];
        agg[gid] = h1[gid] * d * d + b1[f];
    }
}

// One wave32 per edge: lane l gathers float4 l of h1[src] (full 512B row per
// wave via global_load_b128) and scatter-adds into agg[dst] with f32 atomics.
__global__ __launch_bounds__(256) void agg1_edge_kernel(const long long* __restrict__ src,
                                                        const long long* __restrict__ dst,
                                                        const float* __restrict__ h1,
                                                        const float* __restrict__ dis,
                                                        float* __restrict__ agg, long long E) {
    long long e = (long long)blockIdx.x * 8 + (threadIdx.x >> 5);
    int lane = threadIdx.x & 31;
    if (e >= E) return;
    int s = (int)src[e], d = (int)dst[e];
    float nrm = dis[s] * dis[d];
    const float4* hp = (const float4*)(h1 + (size_t)s * HID);
    float4 v = hp[lane];
    float* ap = agg + (size_t)d * HID + lane * 4;
    atomicAdd(ap + 0, v.x * nrm);
    atomicAdd(ap + 1, v.y * nrm);
    atomicAdd(ap + 2, v.z * nrm);
    atomicAdd(ap + 3, v.w * nrm);
}

// ReLU + convert to f16 (input for GEMM2).
__global__ __launch_bounds__(256) void relu_f16_kernel(const float* __restrict__ agg,
                                                       _Float16* __restrict__ hf, int n) {
    int gid = blockIdx.x * blockDim.x + threadIdx.x;
    if (gid < n * HID) hf[gid] = (_Float16)fmaxf(agg[gid], 0.0f);
}

// ---------------------------------------------------------------------------
// GEMM2: h2[N,16] = hf[N,128] @ W2[128,16]. Block = 8 waves, each wave owns
// its own 16-row tile (block covers 128 rows). W2 staged TRANSPOSED in LDS;
// A fragments come straight from global f16 as 16-byte vector loads.
// ---------------------------------------------------------------------------
__global__ __launch_bounds__(256) void gemm2_kernel(const _Float16* __restrict__ hf,
                                                    const float* __restrict__ W2,
                                                    float* __restrict__ h2, int n) {
    __shared__ _Float16 wst[COUT * LDK]; // transposed W2: wst[n*LDK + k], ~4.3 KB
    const int tid = threadIdx.x;
    for (int i = tid; i < HID * COUT; i += 256) {
        int k = i >> 4, c = i & 15;              // W2 is [K=128][N=16] row-major
        wst[c * LDK + k] = (_Float16)W2[i];
    }
    __syncthreads();

    const int wave = tid >> 5;
    const int lane = tid & 31;
    const int rowTile = (blockIdx.x * 8 + wave) * 16;
    if (rowTile >= n) return;             // n % 16 == 0, so full tiles only

    const int m  = lane & 15;
    const int nc = lane & 15;
    const int kb = (lane >> 4) * 8;

    const _Float16* arow = hf + (size_t)(rowTile + m) * HID;
    const _Float16* bcol = &wst[nc * LDK];

    v8f acc = {};
#pragma unroll
    for (int k0 = 0; k0 < HID; k0 += 32) {
        v16h a = frag16(arow + k0 + kb, arow + k0 + 16 + kb);
        v16h b = frag16(bcol + k0 + kb, bcol + k0 + 16 + kb);
        acc = __builtin_amdgcn_wmma_f32_16x16x32_f16(false, a, false, b,
                                                     (short)0, acc, false, false);
    }
#pragma unroll
    for (int r = 0; r < 8; ++r) {
        int mr = r + (lane >> 4) * 8;
        h2[(size_t)(rowTile + mr) * COUT + nc] = acc[r];
    }
}

// ---------------------------------------------------------------------------
// Aggregation layer 2 into d_out (16 features).
// ---------------------------------------------------------------------------
__global__ __launch_bounds__(256) void out_init_kernel(const float* __restrict__ h2,
                                                       const float* __restrict__ dis,
                                                       const float* __restrict__ b2,
                                                       float* __restrict__ out, int n) {
    int gid = blockIdx.x * blockDim.x + threadIdx.x;
    if (gid < n * COUT) {
        int i = gid >> 4, f = gid & 15;
        float d = dis[i];
        out[gid] = h2[gid] * d * d + b2[f];
    }
}

__global__ __launch_bounds__(256) void agg2_edge_kernel(const long long* __restrict__ src,
                                                        const long long* __restrict__ dst,
                                                        const float* __restrict__ h2,
                                                        const float* __restrict__ dis,
                                                        float* __restrict__ out, long long E) {
    long long gid = (long long)blockIdx.x * blockDim.x + threadIdx.x;
    if (gid >= E * COUT) return;
    long long e = gid >> 4;
    int f = (int)(gid & 15);
    int s = (int)src[e], d = (int)dst[e];
    float nrm = dis[s] * dis[d];
    atomicAdd(&out[(size_t)d * COUT + f], h2[(size_t)s * COUT + f] * nrm);
}

// ---------------------------------------------------------------------------
extern "C" void kernel_launch(void* const* d_in, const int* in_sizes, int n_in,
                              void* d_out, int out_size, void* d_ws, size_t ws_size,
                              hipStream_t stream) {
    const float*     x  = (const float*)d_in[0];
    const long long* ei = (const long long*)d_in[1];
    const float*     W1 = (const float*)d_in[2];
    const float*     b1 = (const float*)d_in[3];
    const float*     W2 = (const float*)d_in[4];
    const float*     b2 = (const float*)d_in[5];

    const int       N = in_sizes[0] / FIN;          // 100000
    const long long E = (long long)in_sizes[1] / 2; // 1600000
    const long long* src = ei;
    const long long* dst = ei + E;

    // Workspace layout (aliased across phases):
    //   [dis: N f32][bufA: N*128 f32 (h1, reused as f16 relu buffer)]
    //   [bufB: N*128 f32 (agg1, reused as h2)]
    char* ws = (char*)d_ws;
    size_t off = 0;
    float* dis = (float*)(ws + off);  off += ((size_t)N * 4 + 255) & ~(size_t)255;
    float* h1  = (float*)(ws + off);  off += (size_t)N * HID * 4;
    float* agg = (float*)(ws + off);  off += (size_t)N * HID * 4;
    _Float16* hf = (_Float16*)h1;     // aliases h1 region (h1 dead after agg1)
    float* h2    = agg;               // aliases agg region (agg dead after relu)
    float* out   = (float*)d_out;

    // 1-3: degrees and dis = deg^-1/2
    deg_init_kernel<<<(N + 255) / 256, 256, 0, stream>>>(dis, N);
    deg_count_kernel<<<(unsigned)((E + 255) / 256), 256, 0, stream>>>(dst, dis, E);
    dis_kernel<<<(N + 255) / 256, 256, 0, stream>>>(dis, N);

    // 4: h1 = x @ W1 (WMMA); each block covers 160 rows, staging W1 once
    gemm1_kernel<<<(N + 16 * TILES_PB - 1) / (16 * TILES_PB), 256, 0, stream>>>(x, W1, h1, N);

    // 5-6: layer-1 aggregation (self-loop + bias folded into init)
    agg1_init_kernel<<<(N * HID + 255) / 256, 256, 0, stream>>>(h1, dis, b1, agg, N);
    agg1_edge_kernel<<<(unsigned)((E + 7) / 8), 256, 0, stream>>>(src, dst, h1, dis, agg, E);

    // 7: ReLU -> f16
    relu_f16_kernel<<<(N * HID + 255) / 256, 256, 0, stream>>>(agg, hf, N);

    // 8: h2 = relu(h) @ W2 (WMMA)
    gemm2_kernel<<<(unsigned)((N / 16 + 7) / 8), 256, 0, stream>>>(hf, W2, h2, N);

    // 9-10: layer-2 aggregation into d_out
    out_init_kernel<<<(N * COUT + 255) / 256, 256, 0, stream>>>(h2, dis, b2, out, N);
    agg2_edge_kernel<<<(unsigned)((E * COUT + 255) / 256), 256, 0, stream>>>(src, dst, h2, dis, out, E);
}
